// LSTMDecoder_10282151707711
// MI455X (gfx1250) — matrix operand order repaired and checked by reading, not compiled, not run on hardware
//
#include <hip/hip_runtime.h>
#include <hip/hip_bf16.h>

// ---------------------------------------------------------------------------
// LSTM decoder with (degenerate M=1) attention, LN, tanh, vocab projection.
// B=64, D=512, V=32000, T derived from out_size.
// All matmuls via v_wmma_f32_16x16x32_bf16 (wave32, 16x16x32 tiles).
// ---------------------------------------------------------------------------

typedef __attribute__((ext_vector_type(16))) __bf16 v16bf;
typedef __attribute__((ext_vector_type(8)))  __bf16 v8bf;
typedef __attribute__((ext_vector_type(8)))  float  v8f;

#define Bsz 64
#define Dd  512
#define Vv  32000

__device__ __forceinline__ float sigmoidf_(float x) {
    return 1.0f / (1.0f + __expf(-x));
}

// ---------------------------------------------------------------------------
// GEMM:  C[64, N] = A[64, K] (bf16, row-major) x Bw[N, K]^T (bf16, row-major)
//        + bias[N]   (C is f32 row-major)
// One wave per 16-wide N tile; the wave computes all four 16-row M tiles,
// reusing the B fragment across 4 WMMAs per 32-deep K chunk.
// blockDim = 256 (8 waves). N must be a multiple of 128.
// ---------------------------------------------------------------------------
__global__ void __launch_bounds__(256)
gemm_bf16_wmma(const __bf16* __restrict__ A, const __bf16* __restrict__ Bw,
               const float* __restrict__ bias, float* __restrict__ C,
               int N, int K)
{
    const int lane  = threadIdx.x & 31;
    const int wave  = threadIdx.x >> 5;
    const int ntile = blockIdx.x * 8 + wave;
    if (ntile * 16 >= N) return;               // wave-uniform: EXEC stays all-1s

    const int half = lane >> 4;                // 0 or 1
    const int rowA = lane & 15;
    const int kA   = half * 8;                 // A frag: K 0..7 / 8..15 (+16)
    const int kB   = half * 8;                 // B frag: K 0..7 / 8..15 (+16)
    const int colN = ntile * 16 + (lane & 15);

    v8f acc0 = {}, acc1 = {}, acc2 = {}, acc3 = {};

    const __bf16* Bp = Bw + (size_t)colN * K + kB;
    const __bf16* Ap = A  + (size_t)rowA * K + kA;

    for (int kk = 0; kk < K; kk += 32) {
        union { v16bf v; v8bf h[2]; } bu;
        bu.h[0] = *(const v8bf*)(Bp + kk);
        bu.h[1] = *(const v8bf*)(Bp + kk + 16);
        v16bf bfrag = bu.v;

        union { v16bf v; v8bf h[2]; } a0, a1, a2, a3;
        const __bf16* p = Ap + kk;
        a0.h[0] = *(const v8bf*)(p);
        a0.h[1] = *(const v8bf*)(p + 16);
        a1.h[0] = *(const v8bf*)(p + 16 * K);
        a1.h[1] = *(const v8bf*)(p + 16 * K + 16);
        a2.h[0] = *(const v8bf*)(p + 32 * K);
        a2.h[1] = *(const v8bf*)(p + 32 * K + 16);
        a3.h[0] = *(const v8bf*)(p + 48 * K);
        a3.h[1] = *(const v8bf*)(p + 48 * K + 16);

        acc0 = __builtin_amdgcn_wmma_f32_16x16x32_bf16(false, a0.v, false, bfrag,
                                                       (short)0, acc0, false, false);
        acc1 = __builtin_amdgcn_wmma_f32_16x16x32_bf16(false, a1.v, false, bfrag,
                                                       (short)0, acc1, false, false);
        acc2 = __builtin_amdgcn_wmma_f32_16x16x32_bf16(false, a2.v, false, bfrag,
                                                       (short)0, acc2, false, false);
        acc3 = __builtin_amdgcn_wmma_f32_16x16x32_bf16(false, a3.v, false, bfrag,
                                                       (short)0, acc3, false, false);
    }

    const float bv = bias ? bias[colN] : 0.0f;
    const int mbase = half * 8;
#pragma unroll
    for (int r = 0; r < 8; ++r) {
        C[(size_t)( 0 + mbase + r) * N + colN] = acc0[r] + bv;
        C[(size_t)(16 + mbase + r) * N + colN] = acc1[r] + bv;
        C[(size_t)(32 + mbase + r) * N + colN] = acc2[r] + bv;
        C[(size_t)(48 + mbase + r) * N + colN] = acc3[r] + bv;
    }
}

// ---------------------------------------------------------------------------
// Pointwise LSTM cell.  gates layout: [i | f | g | o] each of width 512.
// Writes new c,h (f32) and stages h (bf16) into xh[:,512:1024] and hc[:,0:512].
// ---------------------------------------------------------------------------
__global__ void __launch_bounds__(256)
lstm_pointwise(const float* __restrict__ gates,
               float* __restrict__ c, float* __restrict__ h,
               __bf16* __restrict__ xh, __bf16* __restrict__ hc)
{
    int idx = blockIdx.x * 256 + threadIdx.x;
    if (idx >= Bsz * Dd) return;
    int b = idx >> 9, d = idx & 511;
    const float* g = gates + (size_t)b * (4 * Dd);
    float iv = g[d], fv = g[Dd + d], gv = g[2 * Dd + d], ov = g[3 * Dd + d];
    float cn = sigmoidf_(fv) * c[idx] + sigmoidf_(iv) * tanhf(gv);
    float hn = sigmoidf_(ov) * tanhf(cn);
    c[idx] = cn;
    h[idx] = hn;
    __bf16 hb = (__bf16)hn;
    hc[(size_t)b * 1024 + d]       = hb;   // pe input: [h | ctx=feats]
    xh[(size_t)b * 1024 + 512 + d] = hb;   // gates input: [x | h]
}

// ---------------------------------------------------------------------------
// LayerNorm + tanh over D=512, one block (256 threads, 2 elems each) per row.
// Writes pe16 (A-matrix for logits GEMM, stride 512) and next-step x into
// xh[:,0:512] (stride 1024).
// ---------------------------------------------------------------------------
__global__ void __launch_bounds__(256)
ln_tanh(const float* __restrict__ pe_raw,
        const float* __restrict__ gamma, const float* __restrict__ beta,
        __bf16* __restrict__ pe16, __bf16* __restrict__ xh)
{
    __shared__ float red[256];
    const int b = blockIdx.x;
    const int t = threadIdx.x;
    float x0 = pe_raw[(size_t)b * Dd + t];
    float x1 = pe_raw[(size_t)b * Dd + 256 + t];

    red[t] = x0 + x1;
    __syncthreads();
    for (int s = 128; s > 0; s >>= 1) {
        if (t < s) red[t] += red[t + s];
        __syncthreads();
    }
    float mu = red[0] * (1.0f / Dd);
    __syncthreads();

    float d0 = x0 - mu, d1 = x1 - mu;
    red[t] = d0 * d0 + d1 * d1;
    __syncthreads();
    for (int s = 128; s > 0; s >>= 1) {
        if (t < s) red[t] += red[t + s];
        __syncthreads();
    }
    float rstd = rsqrtf(red[0] * (1.0f / Dd) + 1e-5f);

    float y0 = tanhf(gamma[t]       * d0 * rstd + beta[t]);
    float y1 = tanhf(gamma[256 + t] * d1 * rstd + beta[256 + t]);
    __bf16 y0b = (__bf16)y0, y1b = (__bf16)y1;
    pe16[(size_t)b * Dd + t]         = y0b;
    pe16[(size_t)b * Dd + 256 + t]   = y1b;
    xh[(size_t)b * 1024 + t]         = y0b;   // next x
    xh[(size_t)b * 1024 + 256 + t]   = y1b;
}

// ---------------------------------------------------------------------------
// One-time conversion / init kernels (run at the top of every launch).
// ---------------------------------------------------------------------------
__global__ void __launch_bounds__(256)
pack_gate_weights(const float* __restrict__ Wih, const float* __restrict__ Whh,
                  __bf16* __restrict__ Wg)
{
    int idx = blockIdx.x * 256 + threadIdx.x;
    if (idx >= 4 * Dd * 2 * Dd) return;            // 2048 x 1024
    int n = idx >> 10, k = idx & 1023;
    float v = (k < Dd) ? Wih[(size_t)n * Dd + k] : Whh[(size_t)n * Dd + (k - Dd)];
    Wg[idx] = (__bf16)v;
}

__global__ void __launch_bounds__(256)
combine_bias(const float* __restrict__ b_ih, const float* __restrict__ b_hh,
             float* __restrict__ bg)
{
    int idx = blockIdx.x * 256 + threadIdx.x;
    if (idx < 4 * Dd) bg[idx] = b_ih[idx] + b_hh[idx];
}

__global__ void __launch_bounds__(256)
convert_f32_bf16(__bf16* __restrict__ dst, const float* __restrict__ src, int n)
{
    int idx = blockIdx.x * 256 + threadIdx.x;
    if (idx < n) dst[idx] = (__bf16)src[idx];
}

// h0 = c0 = feats ; x0 = emb[0] (broadcast) ; hc[:,512:] = ctx = feats (constant,
// exact because the attention memory has a single slot -> softmax == 1).
__global__ void __launch_bounds__(256)
init_state(const float* __restrict__ feats, const float* __restrict__ emb,
           float* __restrict__ h, float* __restrict__ c,
           __bf16* __restrict__ xh, __bf16* __restrict__ hc)
{
    int idx = blockIdx.x * 256 + threadIdx.x;
    if (idx >= Bsz * Dd) return;
    int b = idx >> 9, d = idx & 511;
    float fv = feats[idx];
    h[idx] = fv;
    c[idx] = fv;
    __bf16 fb = (__bf16)fv;
    xh[(size_t)b * 1024 + d]       = (__bf16)emb[d];   // x0 = emb row 0
    xh[(size_t)b * 1024 + 512 + d] = fb;               // h0
    hc[(size_t)b * 1024 + 512 + d] = fb;               // ctx = feats, fixed
}

// ---------------------------------------------------------------------------
// Host-side orchestration
// ---------------------------------------------------------------------------
static inline size_t align256(size_t x) { return (x + 255) & ~(size_t)255; }

extern "C" void kernel_launch(void* const* d_in, const int* in_sizes, int n_in,
                              void* d_out, int out_size, void* d_ws, size_t ws_size,
                              hipStream_t stream)
{
    const float* feats      = (const float*)d_in[0];
    const float* W_ih       = (const float*)d_in[1];
    const float* W_hh       = (const float*)d_in[2];
    const float* b_ih       = (const float*)d_in[3];
    const float* b_hh       = (const float*)d_in[4];
    const float* W_hc       = (const float*)d_in[5];
    const float* b_hc       = (const float*)d_in[6];
    const float* ln_gamma   = (const float*)d_in[7];
    const float* ln_beta    = (const float*)d_in[8];
    const float* emb        = (const float*)d_in[9];
    const float* vocab_bias = (const float*)d_in[10];
    float*       out        = (float*)d_out;

    const int T = out_size / (Bsz * Vv);

    // --- workspace carve-up (all 256B aligned) ---
    char* ws = (char*)d_ws;
    size_t off = 0;
    __bf16* Wg     = (__bf16*)(ws + off); off = align256(off + (size_t)2048 * 1024 * sizeof(__bf16));
    float*  bg     = (float*) (ws + off); off = align256(off + (size_t)2048 * sizeof(float));
    __bf16* Whc16  = (__bf16*)(ws + off); off = align256(off + (size_t)512 * 1024 * sizeof(__bf16));
    __bf16* emb16  = (__bf16*)(ws + off); off = align256(off + (size_t)Vv * Dd * sizeof(__bf16));
    __bf16* xh     = (__bf16*)(ws + off); off = align256(off + (size_t)Bsz * 1024 * sizeof(__bf16));
    __bf16* hc     = (__bf16*)(ws + off); off = align256(off + (size_t)Bsz * 1024 * sizeof(__bf16));
    float*  gates  = (float*) (ws + off); off = align256(off + (size_t)Bsz * 2048 * sizeof(float));
    float*  pe_raw = (float*) (ws + off); off = align256(off + (size_t)Bsz * Dd * sizeof(float));
    __bf16* pe16   = (__bf16*)(ws + off); off = align256(off + (size_t)Bsz * Dd * sizeof(__bf16));
    float*  hbuf   = (float*) (ws + off); off = align256(off + (size_t)Bsz * Dd * sizeof(float));
    float*  cbuf   = (float*) (ws + off); off = align256(off + (size_t)Bsz * Dd * sizeof(float));
    (void)ws_size; (void)n_in; (void)in_sizes;

    // --- one-time weight conversion + state init ---
    {
        int n = 2048 * 1024;
        pack_gate_weights<<<(n + 255) / 256, 256, 0, stream>>>(W_ih, W_hh, Wg);
        combine_bias<<<(2048 + 255) / 256, 256, 0, stream>>>(b_ih, b_hh, bg);
        n = 512 * 1024;
        convert_f32_bf16<<<(n + 255) / 256, 256, 0, stream>>>(Whc16, W_hc, n);
        n = Vv * Dd;
        convert_f32_bf16<<<(n + 255) / 256, 256, 0, stream>>>(emb16, emb, n);
        n = Bsz * Dd;
        init_state<<<(n + 255) / 256, 256, 0, stream>>>(feats, emb, hbuf, cbuf, xh, hc);
    }

    // --- time steps ---
    for (int t = 0; t < T; ++t) {
        // gates = [x|h] @ [W_ih|W_hh]^T + (b_ih+b_hh)     (M=64, N=2048, K=1024)
        gemm_bf16_wmma<<<2048 / 128, 256, 0, stream>>>(xh, Wg, bg, gates, 2048, 1024);

        // LSTM cell pointwise; stage h into xh/hc
        lstm_pointwise<<<(Bsz * Dd + 255) / 256, 256, 0, stream>>>(gates, cbuf, hbuf, xh, hc);

        // pe_raw = [h|feats] @ W_hc^T + b_hc              (M=64, N=512, K=1024)
        gemm_bf16_wmma<<<512 / 128, 256, 0, stream>>>(hc, Whc16, b_hc, pe_raw, 512, 1024);

        // LayerNorm + tanh; stage pe into pe16 and next-step x
        ln_tanh<<<Bsz, 256, 0, stream>>>(pe_raw, ln_gamma, ln_beta, pe16, xh);

        // logits = pe @ emb^T + vocab_bias                (M=64, N=32000, K=512)
        gemm_bf16_wmma<<<Vv / 128, 256, 0, stream>>>(pe16, emb16, vocab_bias,
                                                     out + (size_t)t * Bsz * Vv,
                                                     Vv, 512);
    }
}
